// AngleModel_13262859010049
// MI455X (gfx1250) — compile-verified, matrix operand order checked by reading, not run on hardware
//
#include <hip/hip_runtime.h>
#include <math.h>

typedef __attribute__((ext_vector_type(16))) _Float16 v16h;
typedef __attribute__((ext_vector_type(8)))  _Float16 v8h;
typedef __attribute__((ext_vector_type(8)))  float    v8f;

#define N_NODES 100000
#define N_EDGES 3200000
#define FEAT    64
#define IN_DIM  6

// ---------- order-preserving float<->int encoding for atomic segment-max ----------
__device__ __forceinline__ int enc_f(float f) {
    int b = __float_as_int(f);
    return b ^ ((b >> 31) & 0x7fffffff);   // involution; monotone over all floats
}
__device__ __forceinline__ float dec_f(int i) {
    return __int_as_float(i ^ ((i >> 31) & 0x7fffffff));
}

// ---------- per-layer scratch init ----------
__global__ __launch_bounds__(256)
void init_layer(float* __restrict__ acc, float* __restrict__ denom, int* __restrict__ amax) {
    int i = blockIdx.x * 256 + threadIdx.x;
    if (i < N_NODES * FEAT) acc[i] = 0.0f;
    if (i < N_NODES) { denom[i] = 0.0f; amax[i] = enc_f(-INFINITY); }
}

// ---------- operand packing: X f32 [N,K] -> f16 [N,KPAD], zero padded ----------
__global__ __launch_bounds__(256)
void pack_x(const float* __restrict__ X, _Float16* __restrict__ Xh, int K, int KPAD) {
    const int i = blockIdx.x * 256 + threadIdx.x;
    if (i >= N_NODES * KPAD) return;
    const int n = i / KPAD, j = i - n * KPAD;
    Xh[i] = (j < K) ? (_Float16)X[(long long)n * K + j] : (_Float16)0.0f;
}

// ---------- weight packing: 4 matrices [K,64] f32 -> transposed f16 [4][64][KPAD] ----------
__global__ __launch_bounds__(256)
void pack_w(const float* __restrict__ W0, const float* __restrict__ W1,
            const float* __restrict__ W2, const float* __restrict__ W3,
            _Float16* __restrict__ Wt, int K, int KPAD) {
    const int i = blockIdx.x * 256 + threadIdx.x;
    if (i >= 4 * FEAT * KPAD) return;
    const int m  = i / (FEAT * KPAD);
    const int r  = i - m * FEAT * KPAD;
    const int c  = r / KPAD;
    const int kk = r - c * KPAD;
    const float* W = (m == 0) ? W0 : (m == 1) ? W1 : (m == 2) ? W2 : W3;
    Wt[i] = (kk < K) ? (_Float16)W[(long long)kk * FEAT + c] : (_Float16)0.0f;
}

// ---------- fused q/k/v/skip projections via WMMA f16 (fp32 accumulate) ----------
// Block = 256 threads = 8 waves; block covers 32 rows (N % 32 == 0 -> no guards).
// wave w: matrix m = w>>1, row tile = w&1. Each wave: 16x64 tile of Y_m = Xh @ W_m + b_m.
// Fragments per ISA 7.12.2: A lane=row, B lane=col; half-wave K-groups {0-7,16-23}/{8-15,24-31}.
// Packed layouts make each fragment exactly two contiguous b128 loads.
template <int KPAD>
__global__ __launch_bounds__(256)
void gemm_qkvs(const _Float16* __restrict__ Xh, const _Float16* __restrict__ Wt,
               const float* __restrict__ b0, const float* __restrict__ b1,
               const float* __restrict__ b2, const float* __restrict__ b3,
               float* __restrict__ Y0, float* __restrict__ Y1,
               float* __restrict__ Y2, float* __restrict__ Y3)
{
    const int tid   = threadIdx.x;
    const int wave  = tid >> 5;
    const int lane  = tid & 31;
    const int m     = wave >> 1;
    const int rowBase = blockIdx.x * 32 + (wave & 1) * 16;

    const float* b = (m == 0) ? b0 : (m == 1) ? b1 : (m == 2) ? b2 : b3;
    float*       Y = (m == 0) ? Y0 : (m == 1) ? Y1 : (m == 2) ? Y2 : Y3;
    const _Float16* Wm = Wt + (long long)m * FEAT * KPAD;

    const int khalf = lane >> 4;
    const int l16   = lane & 31 & 15;
    const int row   = rowBase + l16;
    const _Float16* xrow = Xh + (long long)row * KPAD + khalf * 8;

    v8f acc[4] = {};
    constexpr int kTiles = KPAD / 32;

#pragma unroll
    for (int kt = 0; kt < kTiles; ++kt) {
        const int k0 = kt * 32;
        const v8h alo = *(const v8h*)(xrow + k0);
        const v8h ahi = *(const v8h*)(xrow + k0 + 16);
        const v16h a = __builtin_shufflevector(alo, ahi,
            0, 1, 2, 3, 4, 5, 6, 7, 8, 9, 10, 11, 12, 13, 14, 15);
#pragma unroll
        for (int ct = 0; ct < 4; ++ct) {
            const _Float16* wcol = Wm + (long long)(ct * 16 + l16) * KPAD + khalf * 8;
            const v8h blo = *(const v8h*)(wcol + k0);
            const v8h bhi = *(const v8h*)(wcol + k0 + 16);
            const v16h bf = __builtin_shufflevector(blo, bhi,
                0, 1, 2, 3, 4, 5, 6, 7, 8, 9, 10, 11, 12, 13, 14, 15);
            acc[ct] = __builtin_amdgcn_wmma_f32_16x16x32_f16(
                false, a, false, bf, (short)0, acc[ct], false, false);
        }
    }

    // D layout: VGPR r, lanes 0-15 -> M = r; lanes 16-31 -> M = r + 8; N = l16.
#pragma unroll
    for (int ct = 0; ct < 4; ++ct) {
        const int col = ct * 16 + l16;
        const float bias = b[col];
#pragma unroll
        for (int r = 0; r < 8; ++r) {
            const int rr = rowBase + r + 8 * khalf;
            Y[(long long)rr * FEAT + col] = acc[ct][r] + bias;
        }
    }
}

// ---------- edge pass 1: attention logits + segment max (one wave per edge) ----------
__global__ __launch_bounds__(256)
void edge_alpha(const int* __restrict__ src, const int* __restrict__ dst,
                const float* __restrict__ ea,
                const float* __restrict__ q, const float* __restrict__ k,
                const float* __restrict__ We,
                float* __restrict__ alpha, int* __restrict__ amax)
{
    const int e    = blockIdx.x * 8 + (threadIdx.x >> 5);
    const int lane = threadIdx.x & 31;
    if (e >= N_EDGES) return;

    const int s = src[e], d = dst[e];
    const float w = ea[e];
    const float* qd = q + (long long)d * FEAT;
    const float* ks = k + (long long)s * FEAT;

    float p = qd[lane]      * (ks[lane]      + w * We[lane])
            + qd[lane + 32] * (ks[lane + 32] + w * We[lane + 32]);
#pragma unroll
    for (int off = 16; off; off >>= 1) p += __shfl_xor(p, off, 32);

    if (lane == 0) {
        const float al = p * 0.125f;            // 1/sqrt(64)
        alpha[e] = al;
        atomicMax(&amax[d], enc_f(al));
    }
}

// ---------- edge pass 2: exp, denom, unnormalized message scatter-add ----------
__global__ __launch_bounds__(256)
void edge_accum(const int* __restrict__ src, const int* __restrict__ dst,
                const float* __restrict__ ea,
                const float* __restrict__ v, const float* __restrict__ We,
                const float* __restrict__ alpha, const int* __restrict__ amax,
                float* __restrict__ denom, float* __restrict__ acc)
{
    const int e    = blockIdx.x * 8 + (threadIdx.x >> 5);
    const int lane = threadIdx.x & 31;
    if (e >= N_EDGES) return;

    const int s = src[e], d = dst[e];
    const float w  = ea[e];
    const float ex = __expf(alpha[e] - dec_f(amax[d]));
    if (lane == 0) atomicAdd(&denom[d], ex);

    const float* vs = v + (long long)s * FEAT;
    float* ad = acc + (long long)d * FEAT;
    atomicAdd(&ad[lane],      (vs[lane]      + w * We[lane])      * ex);
    atomicAdd(&ad[lane + 32], (vs[lane + 32] + w * We[lane + 32]) * ex);
}

// ---------- node finalize: softmax-normalize, skip connection, relu ----------
__global__ __launch_bounds__(256)
void node_finalize(const float* __restrict__ acc, const float* __restrict__ denom,
                   const float* __restrict__ skip, float* __restrict__ h)
{
    const int i = blockIdx.x * 256 + threadIdx.x;
    if (i >= N_NODES * FEAT) return;
    const int n = i >> 6;
    const float val = acc[i] / (denom[n] + 1e-16f) + skip[i];
    h[i] = fmaxf(val, 0.0f);
}

// ---------- head: 64->3 fc, L2-normalize*10, sentinel masks, drop last row ----------
__global__ __launch_bounds__(256)
void head(const float* __restrict__ h, const float* __restrict__ x,
          const float* __restrict__ Wfc, const float* __restrict__ bfc,
          float* __restrict__ out)
{
    const int n = blockIdx.x * 256 + threadIdx.x;
    if (n >= N_NODES - 1) return;      // reference returns out[:-1]

    float o0 = bfc[0], o1 = bfc[1], o2 = bfc[2];
    const float* hn = h + (long long)n * FEAT;
#pragma unroll 8
    for (int c = 0; c < FEAT; ++c) {
        const float hv = hn[c];        // already relu'd (extra relu is idempotent)
        o0 += hv * Wfc[c * 3 + 0];
        o1 += hv * Wfc[c * 3 + 1];
        o2 += hv * Wfc[c * 3 + 2];
    }
    const float nrm = sqrtf(o0 * o0 + o1 * o1 + o2 * o2);
    const float inv = 10.0f / fmaxf(nrm, 1e-12f);
    o0 *= inv; o1 *= inv; o2 *= inv;
    if (x[(long long)n * IN_DIM + 3] == -1.0f) o0 -= 10.0f;
    if (x[(long long)n * IN_DIM + 5] ==  1.0f) o2 -= 10.0f;
    out[n * 3 + 0] = o0; out[n * 3 + 1] = o1; out[n * 3 + 2] = o2;
}

extern "C" void kernel_launch(void* const* d_in, const int* in_sizes, int n_in,
                              void* d_out, int out_size, void* d_ws, size_t ws_size,
                              hipStream_t stream) {
    (void)in_sizes; (void)n_in; (void)out_size; (void)ws_size;

    const float* x   = (const float*)d_in[0];
    const int*   ei  = (const int*)d_in[1];
    const float* ea  = (const float*)d_in[2];
    const float *Wq1 = (const float*)d_in[3],  *bq1 = (const float*)d_in[4];
    const float *Wk1 = (const float*)d_in[5],  *bk1 = (const float*)d_in[6];
    const float *Wv1 = (const float*)d_in[7],  *bv1 = (const float*)d_in[8];
    const float *We1 = (const float*)d_in[9];
    const float *Ws1 = (const float*)d_in[10], *bs1 = (const float*)d_in[11];
    const float *Wq2 = (const float*)d_in[12], *bq2 = (const float*)d_in[13];
    const float *Wk2 = (const float*)d_in[14], *bk2 = (const float*)d_in[15];
    const float *Wv2 = (const float*)d_in[16], *bv2 = (const float*)d_in[17];
    const float *We2 = (const float*)d_in[18];
    const float *Ws2 = (const float*)d_in[19], *bs2 = (const float*)d_in[20];
    const float *Wfc = (const float*)d_in[21], *bfc = (const float*)d_in[22];

    const int* src = ei;
    const int* dst = ei + N_EDGES;

    // workspace layout — per-layer hot set ~115 MB, fits the 192 MB L2
    const long long N64 = (long long)N_NODES * FEAT;
    float* q     = (float*)d_ws;
    float* k     = q + N64;
    float* v     = k + N64;
    float* skip  = v + N64;
    float* acc   = skip + N64;
    float* h     = acc + N64;
    float* denom = h + N64;
    int*   amax  = (int*)(denom + N_NODES);
    float* alpha = (float*)(amax + N_NODES);
    _Float16* Xh = (_Float16*)(alpha + N_EDGES);     // 16B-aligned (offset is multiple of 16)
    _Float16* Wt = Xh + N64;                          // 4 * 64 * 64 halfs

    const dim3 b256(256);
    const int gInit  = (N_NODES * FEAT + 255) / 256;  // 25000
    const int gGemm  = (N_NODES + 31) / 32;           // 3125
    const int gEdge  = (N_EDGES + 7) / 8;             // 400000 (wave per edge)
    const int gNode  = (N_NODES * FEAT + 255) / 256;
    const int gHead  = (N_NODES - 1 + 255) / 256;
    const int gPackX1 = (N_NODES * 32 + 255) / 256;
    const int gPackX2 = (N_NODES * 64 + 255) / 256;
    const int gPackW1 = (4 * FEAT * 32 + 255) / 256;
    const int gPackW2 = (4 * FEAT * 64 + 255) / 256;

    // ---- layer 1 (K = 6 zero-padded to one 32-wide WMMA K-tile) ----
    init_layer<<<gInit, b256, 0, stream>>>(acc, denom, amax);
    pack_x<<<gPackX1, b256, 0, stream>>>(x, Xh, IN_DIM, 32);
    pack_w<<<gPackW1, b256, 0, stream>>>(Wq1, Wk1, Wv1, Ws1, Wt, IN_DIM, 32);
    gemm_qkvs<32><<<gGemm, b256, 0, stream>>>(Xh, Wt, bq1, bk1, bv1, bs1, q, k, v, skip);
    edge_alpha<<<gEdge, b256, 0, stream>>>(src, dst, ea, q, k, We1, alpha, amax);
    edge_accum<<<gEdge, b256, 0, stream>>>(src, dst, ea, v, We1, alpha, amax, denom, acc);
    node_finalize<<<gNode, b256, 0, stream>>>(acc, denom, skip, h);

    // ---- layer 2 (K = 64 -> two WMMA K-tiles, fully unrolled) ----
    init_layer<<<gInit, b256, 0, stream>>>(acc, denom, amax);
    pack_x<<<gPackX2, b256, 0, stream>>>(h, Xh, FEAT, 64);
    pack_w<<<gPackW2, b256, 0, stream>>>(Wq2, Wk2, Wv2, Ws2, Wt, FEAT, 64);
    gemm_qkvs<64><<<gGemm, b256, 0, stream>>>(Xh, Wt, bq2, bk2, bv2, bs2, q, k, v, skip);
    edge_alpha<<<gEdge, b256, 0, stream>>>(src, dst, ea, q, k, We2, alpha, amax);
    edge_accum<<<gEdge, b256, 0, stream>>>(src, dst, ea, v, We2, alpha, amax, denom, acc);
    node_finalize<<<gNode, b256, 0, stream>>>(acc, denom, skip, h);

    // ---- head ----
    head<<<gHead, b256, 0, stream>>>(h, x, Wfc, bfc, (float*)d_out);
}
